// Decoder_12816182411830
// MI455X (gfx1250) — matrix operand order repaired and checked by reading, not compiled, not run on hardware
//
#include <hip/hip_runtime.h>
#include <cstdint>
#include <cstddef>

// ---------------- problem constants ----------------
#define Bsz 256
#define Lsz 400
#define Esz 128
#define Hsz 256
#define V4sz 50004
#define H2 (2*Hsz)   // 512
#define H3 (3*Hsz)   // 768
#define PGEN_DIM (4*Hsz + Esz) // 1152

typedef __attribute__((ext_vector_type(16))) _Float16 v16h;
typedef __attribute__((ext_vector_type(8)))  _Float16 v8h;
typedef __attribute__((ext_vector_type(4)))  _Float16 v4h;
typedef __attribute__((ext_vector_type(8)))  float    v8f;

__device__ __forceinline__ float sigf(float x) { return 1.0f / (1.0f + __expf(-x)); }
__device__ __forceinline__ float dot4(float4 a, float4 b) {
    return a.x*b.x + a.y*b.y + a.z*b.z + a.w*b.w;
}

// ---------------- 1) f32 -> f16 weight conversion (out2_w) ----------------
__global__ void k_cvt_f16(const float* __restrict__ src, _Float16* __restrict__ dst, int n4) {
    int i = blockIdx.x * blockDim.x + threadIdx.x;
    if (i >= n4) return;
    float4 v = reinterpret_cast<const float4*>(src)[i];
    v4h h;
    h[0] = (_Float16)v.x; h[1] = (_Float16)v.y; h[2] = (_Float16)v.z; h[3] = (_Float16)v.w;
    reinterpret_cast<v4h*>(dst)[i] = h;
}

// ---------------- 2) LSTM single step ----------------
// grid = B blocks of H threads; thread (b,k) computes gate lane k.
__global__ void k_lstm(const float* __restrict__ x,   // [B,E]
                       const float* __restrict__ h0,  // [B,H]
                       const float* __restrict__ c0,  // [B,H]
                       const float* __restrict__ Wih, // [4H,E]
                       const float* __restrict__ Whh, // [4H,H]
                       const float* __restrict__ bih, const float* __restrict__ bhh,
                       float* __restrict__ h_new, float* __restrict__ c_new) {
    int b = blockIdx.x, k = threadIdx.x;
    const float4* xb = reinterpret_cast<const float4*>(x + (size_t)b * Esz);
    const float4* hb = reinterpret_cast<const float4*>(h0 + (size_t)b * Hsz);
    float acc[4];
#pragma unroll
    for (int g = 0; g < 4; ++g) {
        int row = g * Hsz + k;
        float a = bih[row] + bhh[row];
        const float4* wi = reinterpret_cast<const float4*>(Wih + (size_t)row * Esz);
#pragma unroll 4
        for (int t = 0; t < Esz/4; ++t) a += dot4(wi[t], xb[t]);
        const float4* wh = reinterpret_cast<const float4*>(Whh + (size_t)row * Hsz);
#pragma unroll 4
        for (int t = 0; t < Hsz/4; ++t) a += dot4(wh[t], hb[t]);
        acc[g] = a;
    }
    float ig = sigf(acc[0]);
    float fg = sigf(acc[1]);
    float gg = tanhf(acc[2]);
    float og = sigf(acc[3]);
    float c1 = fg * c0[(size_t)b * Hsz + k] + ig * gg;
    float h1 = og * tanhf(c1);
    h_new[(size_t)b * Hsz + k] = h1;
    c_new[(size_t)b * Hsz + k] = c1;
}

// ---------------- 3) ws_ = s @ ws_w.T + ws_b   (s = [h1|c1]) ----------------
// grid = (2, B), block 256; thread -> (b, d)
__global__ void k_wslin(const float* __restrict__ h1, const float* __restrict__ c1,
                        const float* __restrict__ wsw, // [2H,2H]
                        const float* __restrict__ wsb,
                        float* __restrict__ out) {     // [B,2H]
    int b = blockIdx.y;
    int d = blockIdx.x * blockDim.x + threadIdx.x;
    const float4* w1 = reinterpret_cast<const float4*>(wsw + (size_t)d * H2);
    const float4* w2 = reinterpret_cast<const float4*>(wsw + (size_t)d * H2 + Hsz);
    const float4* hr = reinterpret_cast<const float4*>(h1 + (size_t)b * Hsz);
    const float4* cr = reinterpret_cast<const float4*>(c1 + (size_t)b * Hsz);
    float a = wsb[d];
#pragma unroll 4
    for (int t = 0; t < Hsz/4; ++t) a += dot4(w1[t], hr[t]);
#pragma unroll 4
    for (int t = 0; t < Hsz/4; ++t) a += dot4(w2[t], cr[t]);
    out[(size_t)b * H2 + d] = a;
}

// ---------------- 4) attention scores e[b,l] ----------------
// one thread per (b,l); streams one 2KB enc_out row per thread.
__global__ void k_escore(const float* __restrict__ enc_out, const float* __restrict__ enc_mask,
                         const float* __restrict__ cov, const int* __restrict__ is_cov,
                         const float* __restrict__ vw, const float* __restrict__ vb,
                         const float* __restrict__ wcw, const float* __restrict__ wcb,
                         const float* __restrict__ wslin, float* __restrict__ e) {
    int idx = blockIdx.x * blockDim.x + threadIdx.x; // B*L
    int b = idx / Lsz, l = idx % Lsz;
    bool uc = (is_cov[0] != 0);
    float cv = uc ? cov[idx] : 0.0f;
    const float4* eo = reinterpret_cast<const float4*>(enc_out + ((size_t)b * Lsz + l) * H2);
    const float4* ws = reinterpret_cast<const float4*>(wslin + (size_t)b * H2);
    const float4* v4 = reinterpret_cast<const float4*>(vw);
    const float4* cw = reinterpret_cast<const float4*>(wcw);
    const float4* cb = reinterpret_cast<const float4*>(wcb);
    float a = vb[0];
    for (int t = 0; t < H2/4; ++t) {
        float4 s = eo[t], w = ws[t], v = v4[t];
        float4 x;
        x.x = s.x + w.x; x.y = s.y + w.y; x.z = s.z + w.z; x.w = s.w + w.w;
        if (uc) {
            float4 c = cw[t], bb = cb[t];
            x.x += cv * c.x + bb.x; x.y += cv * c.y + bb.y;
            x.z += cv * c.z + bb.z; x.w += cv * c.w + bb.w;
        }
        a += v.x * tanhf(x.x) + v.y * tanhf(x.y) + v.z * tanhf(x.z) + v.w * tanhf(x.w);
    }
    a += (1.0f - enc_mask[idx]) * -1e9f;
    e[idx] = a;
}

// ---------------- 5) softmax over L, attn/coverage out, context ----------------
// grid = B, block 256
__global__ void k_attn_ctx(const float* __restrict__ e, const float* __restrict__ enc_out,
                           const float* __restrict__ cov, const int* __restrict__ is_cov,
                           float* __restrict__ attn_out, float* __restrict__ cov_out,
                           float* __restrict__ ctx) {
    __shared__ float red[256];
    __shared__ float aL[Lsz];
    int b = blockIdx.x, t = threadIdx.x;
    const float* eb = e + (size_t)b * Lsz;
    float m = -3.4e38f;
    for (int l = t; l < Lsz; l += 256) m = fmaxf(m, eb[l]);
    red[t] = m; __syncthreads();
    for (int s = 128; s > 0; s >>= 1) { if (t < s) red[t] = fmaxf(red[t], red[t + s]); __syncthreads(); }
    m = red[0]; __syncthreads();
    float sum = 0.0f;
    for (int l = t; l < Lsz; l += 256) { float x = __expf(eb[l] - m); aL[l] = x; sum += x; }
    red[t] = sum; __syncthreads();
    for (int s = 128; s > 0; s >>= 1) { if (t < s) red[t] += red[t + s]; __syncthreads(); }
    float inv = 1.0f / red[0];
    bool uc = (is_cov[0] != 0);
    __syncthreads();
    for (int l = t; l < Lsz; l += 256) {
        float a = aL[l] * inv;
        aL[l] = a;
        attn_out[(size_t)b * Lsz + l] = a;
        cov_out[(size_t)b * Lsz + l] = cov[(size_t)b * Lsz + l] + (uc ? a : 0.0f);
    }
    __syncthreads();
    // context[b,d] = sum_l attn[l] * enc_out[b,l,d]; thread handles d=t and d=t+256
    const float* base = enc_out + (size_t)b * Lsz * H2;
    float a0 = 0.0f, a1 = 0.0f;
    for (int l = 0; l < Lsz; ++l) {
        float a = aL[l];
        a0 += a * base[(size_t)l * H2 + t];
        a1 += a * base[(size_t)l * H2 + Hsz + t];
    }
    ctx[(size_t)b * H2 + t] = a0;
    ctx[(size_t)b * H2 + Hsz + t] = a1;
}

// ---------------- 6) hid = [context|h1] @ out1_w.T + b  -> f16 ----------------
// grid = B, block H
__global__ void k_hid(const float* __restrict__ ctx, const float* __restrict__ h1,
                      const float* __restrict__ w,  // [H,3H]
                      const float* __restrict__ bias,
                      _Float16* __restrict__ hid16) {
    int b = blockIdx.x, j = threadIdx.x;
    const float4* wr = reinterpret_cast<const float4*>(w + (size_t)j * H3);
    const float4* cr = reinterpret_cast<const float4*>(ctx + (size_t)b * H2);
    const float4* hr = reinterpret_cast<const float4*>(h1 + (size_t)b * Hsz);
    float a = bias[j];
#pragma unroll 4
    for (int t = 0; t < H2/4; ++t) a += dot4(wr[t], cr[t]);
#pragma unroll 4
    for (int t = 0; t < Hsz/4; ++t) a += dot4(wr[H2/4 + t], hr[t]);
    hid16[(size_t)b * Hsz + j] = (_Float16)a;
}

// ---------------- 7) pgen ----------------
// grid = B, block 128
__global__ void k_pgen(const float* __restrict__ ctx, const float* __restrict__ h1,
                       const float* __restrict__ c1, const float* __restrict__ x,
                       const float* __restrict__ pw_, const float* __restrict__ pb_,
                       float* __restrict__ pgen) {
    __shared__ float red[128];
    int b = blockIdx.x, t = threadIdx.x;
    float a = 0.0f;
    for (int i = t; i < H2; i += 128)  a += ctx[(size_t)b * H2 + i] * pw_[i];
    for (int i = t; i < Hsz; i += 128) a += h1[(size_t)b * Hsz + i] * pw_[H2 + i];
    for (int i = t; i < Hsz; i += 128) a += c1[(size_t)b * Hsz + i] * pw_[H2 + Hsz + i];
    for (int i = t; i < Esz; i += 128) a += x[(size_t)b * Esz + i] * pw_[4*Hsz + i];
    red[t] = a; __syncthreads();
    for (int s = 64; s > 0; s >>= 1) { if (t < s) red[t] += red[t + s]; __syncthreads(); }
    if (t == 0) pgen[b] = sigf(red[0] + pb_[0]);
}

// ---------------- 8) logits = hid @ out2_w.T + b2  (WMMA f16 -> f32) ----------------
// grid = (ceil(V4/64), B/16), block 128 (4 waves); wave computes a 16x16 tile.
__global__ void k_logits_wmma(const _Float16* __restrict__ A,  // hid  [B,H] f16
                              const _Float16* __restrict__ Bw, // out2 [V4,H] f16
                              const float* __restrict__ bias,  // [V4]
                              float* __restrict__ logits) {    // [B,V4]
    int wave = threadIdx.x >> 5;
    int lane = threadIdx.x & 31;
    int hi   = lane >> 4;    // half-wave select (ISA 16-bit A/B layout)
    int lm   = lane & 15;
    int mbase = blockIdx.y * 16;
    int nbase = blockIdx.x * 64 + wave * 16;
    int n  = nbase + lm;
    int nc = (n < V4sz) ? n : (V4sz - 1);  // clamp loads at the ragged edge

    v8f c;
    float bb = (n < V4sz) ? bias[n] : 0.0f;
#pragma unroll
    for (int r = 0; r < 8; ++r) c[r] = bb;

    const _Float16* arow = A  + (size_t)(mbase + lm) * Hsz;
    const _Float16* brow = Bw + (size_t)nc * Hsz;

#pragma unroll
    for (int kt = 0; kt < Hsz / 32; ++kt) {
        int k0 = kt * 32;
        // A 16x32 f16 fragment: row m=lm, lanes 0-15 K=k0..k0+15, lanes 16-31 K=k0+16..k0+31
        v16h af, bf;
        af = *reinterpret_cast<const v16h*>(arow + k0 + 16 * hi);
        // B 32x16 f16 fragment: col n=lm, lanes 0-15 K=k0..k0+15, lanes 16-31 K=k0+16..k0+31
        bf = *reinterpret_cast<const v16h*>(brow + k0 + 16 * hi);
        c = __builtin_amdgcn_wmma_f32_16x16x32_f16(
                false, af, false, bf, (short)0, c, false, false);
    }

    if (n < V4sz) {
#pragma unroll
        for (int r = 0; r < 8; ++r) {
            int m = mbase + r + 8 * hi;  // C/D layout: VGPR r -> row r (+8 for upper half-wave)
            logits[(size_t)m * V4sz + n] = c[r];
        }
    }
}

// ---------------- 9) row max / sumexp over V4 ----------------
// grid = B, block 256
__global__ void k_rowstat(const float* __restrict__ logits, float* __restrict__ rmax,
                          float* __restrict__ rsum) {
    __shared__ float red[256];
    int b = blockIdx.x, t = threadIdx.x;
    const float* row = logits + (size_t)b * V4sz;
    float m = -3.4e38f;
    for (int v = t; v < V4sz; v += 256) m = fmaxf(m, row[v]);
    red[t] = m; __syncthreads();
    for (int s = 128; s > 0; s >>= 1) { if (t < s) red[t] = fmaxf(red[t], red[t + s]); __syncthreads(); }
    m = red[0]; __syncthreads();
    float sum = 0.0f;
    for (int v = t; v < V4sz; v += 256) sum += __expf(row[v] - m);
    red[t] = sum; __syncthreads();
    for (int s = 128; s > 0; s >>= 1) { if (t < s) red[t] += red[t + s]; __syncthreads(); }
    if (t == 0) { rmax[b] = m; rsum[b] = red[0]; }
}

// ---------------- 10) pw = softmax(logits) * pgen ----------------
// grid = (ceil(V4/256), B)
__global__ void k_pw(const float* __restrict__ logits, const float* __restrict__ rmax,
                     const float* __restrict__ rsum, const float* __restrict__ pgen,
                     float* __restrict__ pw) {
    int b = blockIdx.y;
    int v = blockIdx.x * blockDim.x + threadIdx.x;
    if (v >= V4sz) return;
    float p = __expf(logits[(size_t)b * V4sz + v] - rmax[b]) * (pgen[b] / rsum[b]);
    pw[(size_t)b * V4sz + v] = p;
}

// ---------------- 11) scatter-add copy mechanism ----------------
__global__ void k_scatter(const long long* __restrict__ enc_ind, const float* __restrict__ attn,
                          const float* __restrict__ pgen, float* __restrict__ pw) {
    int idx = blockIdx.x * blockDim.x + threadIdx.x;
    if (idx >= Bsz * Lsz) return;
    int b = idx / Lsz;
    long long v = enc_ind[idx];
    float add = attn[idx] * (1.0f - pgen[b]);
    atomicAdd(pw + (size_t)b * V4sz + (size_t)v, add);
}

// ---------------- host launch ----------------
extern "C" void kernel_launch(void* const* d_in, const int* in_sizes, int n_in,
                              void* d_out, int out_size, void* d_ws, size_t ws_size,
                              hipStream_t stream) {
    (void)in_sizes; (void)n_in; (void)out_size; (void)ws_size;
    const float* dec_inp  = (const float*)d_in[0];
    const float* enc_out  = (const float*)d_in[1];
    const float* enc_mask = (const float*)d_in[2];
    const float* h_in     = (const float*)d_in[3];
    const float* c_in     = (const float*)d_in[4];
    const long long* enc_ind = (const long long*)d_in[5];
    const float* coverage = (const float*)d_in[6];
    const int*   is_cov   = (const int*)d_in[7];
    const float* W_ih     = (const float*)d_in[8];
    const float* W_hh     = (const float*)d_in[9];
    const float* b_ih     = (const float*)d_in[10];
    const float* b_hh     = (const float*)d_in[11];
    const float* v_w      = (const float*)d_in[12];
    const float* v_b      = (const float*)d_in[13];
    const float* wc_w     = (const float*)d_in[14];
    const float* wc_b     = (const float*)d_in[15];
    const float* ws_w     = (const float*)d_in[16];
    const float* ws_b     = (const float*)d_in[17];
    const float* out1_w   = (const float*)d_in[18];
    const float* out1_b   = (const float*)d_in[19];
    const float* out2_w   = (const float*)d_in[20];
    const float* out2_b   = (const float*)d_in[21];
    const float* pgen_w   = (const float*)d_in[22];
    const float* pgen_b   = (const float*)d_in[23];

    // output layout: pw | attn | coverage_new | h_new | c_new
    float* out   = (float*)d_out;
    float* pw    = out;
    float* attn  = pw + (size_t)Bsz * V4sz;
    float* covo  = attn + (size_t)Bsz * Lsz;
    float* h_new = covo + (size_t)Bsz * Lsz;
    float* c_new = h_new + (size_t)Bsz * Hsz;

    // workspace layout (bytes, all 32B-aligned)
    char* ws = (char*)d_ws;
    float*    ws_lin = (float*)(ws + 0);                              // [B,2H]
    float*    ctx    = (float*)(ws + 524288);                         // [B,2H]
    float*    e      = (float*)(ws + 1048576);                        // [B,L]
    _Float16* hid16  = (_Float16*)(ws + 1458176);                     // [B,H]
    float*    pgen   = (float*)(ws + 1589248);                        // [B]
    float*    rmax   = (float*)(ws + 1590272);                        // [B]
    float*    rsum   = (float*)(ws + 1591296);                        // [B]
    float*    logits = (float*)(ws + 1592320);                        // [B,V4]
    _Float16* w2h    = (_Float16*)(ws + 1592320 + (size_t)Bsz*V4sz*4);// [V4,H] f16

    // 1) convert out2_w to f16
    {
        int n4 = (V4sz * Hsz) / 4;
        k_cvt_f16<<<dim3((n4 + 255) / 256), dim3(256), 0, stream>>>(out2_w, w2h, n4);
    }
    // 2) LSTM
    k_lstm<<<dim3(Bsz), dim3(Hsz), 0, stream>>>(dec_inp, h_in, c_in, W_ih, W_hh, b_ih, b_hh,
                                                h_new, c_new);
    // 3) ws_ linear
    k_wslin<<<dim3(2, Bsz), dim3(256), 0, stream>>>(h_new, c_new, ws_w, ws_b, ws_lin);
    // 4) attention scores
    k_escore<<<dim3((Bsz * Lsz) / 256), dim3(256), 0, stream>>>(
        enc_out, enc_mask, coverage, is_cov, v_w, v_b, wc_w, wc_b, ws_lin, e);
    // 5) softmax + attn/coverage out + context
    k_attn_ctx<<<dim3(Bsz), dim3(256), 0, stream>>>(e, enc_out, coverage, is_cov, attn, covo, ctx);
    // 6) hid (f16)
    k_hid<<<dim3(Bsz), dim3(Hsz), 0, stream>>>(ctx, h_new, out1_w, out1_b, hid16);
    // 7) pgen
    k_pgen<<<dim3(Bsz), dim3(128), 0, stream>>>(ctx, h_new, c_new, dec_inp, pgen_w, pgen_b, pgen);
    // 8) big GEMM via WMMA
    k_logits_wmma<<<dim3((V4sz + 63) / 64, Bsz / 16), dim3(128), 0, stream>>>(
        hid16, w2h, out2_b, logits);
    // 9) softmax row stats over V4
    k_rowstat<<<dim3(Bsz), dim3(256), 0, stream>>>(logits, rmax, rsum);
    // 10) pw = softmax * pgen
    k_pw<<<dim3((V4sz + 255) / 256, Bsz), dim3(256), 0, stream>>>(logits, rmax, rsum, pgen, pw);
    // 11) scatter-add (copy mechanism)
    k_scatter<<<dim3((Bsz * Lsz + 255) / 256), dim3(256), 0, stream>>>(enc_ind, attn, pgen, pw);
}
